// CausalSelfAttention_72241349919284
// MI455X (gfx1250) — matrix operand order repaired and checked by reading, not compiled, not run on hardware
//
#include <hip/hip_runtime.h>
#include <math.h>

// ---------------------------------------------------------------------------
// CDNA5 (gfx1250) causal self-attention, bf16 WMMA everywhere.
// B=2, T=2048, C=1024, H=16, hd=64.
// Round 2: register double-buffering so loads overlap the WMMA pipe,
// 32x64 wave tiles in the GEMMs, early-issued fragment loads in flash attn.
// ---------------------------------------------------------------------------

typedef __bf16 bf16;
typedef __attribute__((ext_vector_type(16))) __bf16 v16bf;
typedef __attribute__((ext_vector_type(8)))  __bf16 v8bf;
typedef __attribute__((ext_vector_type(4)))  __bf16 v4bf;
typedef __attribute__((ext_vector_type(8)))  float   v8f;
typedef __attribute__((ext_vector_type(4)))  float   v4f;

#define DI __device__ __forceinline__

static constexpr int B_  = 2;
static constexpr int T_  = 2048;
static constexpr int C_  = 1024;
static constexpr int H_  = 16;
static constexpr int HD_ = 64;

// ---- WMMA fragment loaders (layouts per CDNA5 ISA §7.12.2) ------
//
// A (16x32 bf16, M x K), row-major source with leading dim ldk:
//   lane = half*16 + row ; element e -> K = 16*(e>=8) + 8*half + (e&7)
DI v16bf load_a_frag(const bf16* __restrict__ base, int ldk, int lane) {
  const int row = lane & 15, half = lane >> 4;
  const bf16* p = base + (size_t)row * ldk + half * 8;
  v8bf c0 = *(const v8bf*)(p);
  v8bf c1 = *(const v8bf*)(p + 16);
  v16bf a;
#pragma unroll
  for (int i = 0; i < 8; ++i) { a[i] = c0[i]; a[i + 8] = c1[i]; }
  return a;
}

// B (32x16 bf16, K x N) with B[k][n] = SRC[n][k], SRC row-major [N, K]:
//   lane = half*16 + n ; element e -> K = 16*(e>=8) + 8*half + (e&7)
DI v16bf load_b_frag(const bf16* __restrict__ base, int ldk, int lane) {
  const int n = lane & 15, half = lane >> 4;
  const bf16* p = base + (size_t)n * ldk + half * 8;
  v8bf c0 = *(const v8bf*)(p);
  v8bf c1 = *(const v8bf*)(p + 16);
  v16bf b;
#pragma unroll
  for (int i = 0; i < 8; ++i) { b[i] = c0[i]; b[i + 8] = c1[i]; }
  return b;
}

DI v8f wmma_bf16(v16bf a, v16bf b, v8f c) {
  return __builtin_amdgcn_wmma_f32_16x16x32_bf16(
      false, a, false, b, (short)0, c, false, false);
}

// ---------------------------------------------------------------------------
// Kernel 0: fp32 -> bf16 cast, 4-wide vectorized.
// ---------------------------------------------------------------------------
__global__ void cast_f32_bf16(const float* __restrict__ src,
                              bf16* __restrict__ dst, int n4) {
  int i = blockIdx.x * blockDim.x + threadIdx.x;
  const int stride = gridDim.x * blockDim.x;
  for (; i < n4; i += stride) {
    v4f f = *(const v4f*)(src + 4 * (size_t)i);
    v4bf o;
#pragma unroll
    for (int j = 0; j < 4; ++j) o[j] = (bf16)f[j];
    *(v4bf*)(dst + 4 * (size_t)i) = o;
  }
}

// ---------------------------------------------------------------------------
// Kernel 1: qkv = X @ Wattn^T + b_attn -> head-major Q/K, transposed V.
// 8 waves/block; each wave owns a 32x64 tile (8 WMMAs per K-step),
// double-buffered fragments. grid = (M/256, 3C/64), block = 256.
// ---------------------------------------------------------------------------
__global__ __launch_bounds__(256) void qkv_gemm(
    const bf16* __restrict__ X,     // [B*T, C]
    const bf16* __restrict__ W,     // [3C, C]
    const float* __restrict__ bias, // [3C]
    bf16* __restrict__ Q,           // [B*H, T, 64]
    bf16* __restrict__ Kh,          // [B*H, T, 64]
    bf16* __restrict__ Vt)          // [B*H, 64, T]
{
  const int lane  = threadIdx.x & 31;
  const int wave  = threadIdx.x >> 5;
  const int mbase = blockIdx.x * 256 + wave * 32;
  const int nbase = blockIdx.y * 64;

  const bf16* Ap = X + (size_t)mbase * C_;
  const bf16* Bp = W + (size_t)nbase * C_;

  v16bf a0 = load_a_frag(Ap, C_, lane);
  v16bf a1 = load_a_frag(Ap + (size_t)16 * C_, C_, lane);
  v16bf b0 = load_b_frag(Bp, C_, lane);
  v16bf b1 = load_b_frag(Bp + (size_t)16 * C_, C_, lane);
  v16bf b2 = load_b_frag(Bp + (size_t)32 * C_, C_, lane);
  v16bf b3 = load_b_frag(Bp + (size_t)48 * C_, C_, lane);

  v8f acc[8] = {};
  for (int kk = 32; kk < C_; kk += 32) {
    // issue next-tile fragment loads before this tile's WMMAs
    v16bf na0 = load_a_frag(Ap + kk, C_, lane);
    v16bf na1 = load_a_frag(Ap + (size_t)16 * C_ + kk, C_, lane);
    v16bf nb0 = load_b_frag(Bp + kk, C_, lane);
    v16bf nb1 = load_b_frag(Bp + (size_t)16 * C_ + kk, C_, lane);
    v16bf nb2 = load_b_frag(Bp + (size_t)32 * C_ + kk, C_, lane);
    v16bf nb3 = load_b_frag(Bp + (size_t)48 * C_ + kk, C_, lane);
    if (kk + 64 < C_) {
      __builtin_prefetch((const void*)(Ap + kk + 64), 0, 0);
      __builtin_prefetch((const void*)(Bp + kk + 64), 0, 0);
    }
    acc[0] = wmma_bf16(a0, b0, acc[0]);
    acc[1] = wmma_bf16(a0, b1, acc[1]);
    acc[2] = wmma_bf16(a0, b2, acc[2]);
    acc[3] = wmma_bf16(a0, b3, acc[3]);
    acc[4] = wmma_bf16(a1, b0, acc[4]);
    acc[5] = wmma_bf16(a1, b1, acc[5]);
    acc[6] = wmma_bf16(a1, b2, acc[6]);
    acc[7] = wmma_bf16(a1, b3, acc[7]);
    a0 = na0; a1 = na1; b0 = nb0; b1 = nb1; b2 = nb2; b3 = nb3;
  }
  acc[0] = wmma_bf16(a0, b0, acc[0]);
  acc[1] = wmma_bf16(a0, b1, acc[1]);
  acc[2] = wmma_bf16(a0, b2, acc[2]);
  acc[3] = wmma_bf16(a0, b3, acc[3]);
  acc[4] = wmma_bf16(a1, b0, acc[4]);
  acc[5] = wmma_bf16(a1, b1, acc[5]);
  acc[6] = wmma_bf16(a1, b2, acc[6]);
  acc[7] = wmma_bf16(a1, b3, acc[7]);

  const int half = lane >> 4, col = lane & 15;
#pragma unroll
  for (int mt = 0; mt < 2; ++mt) {
#pragma unroll
    for (int nt = 0; nt < 4; ++nt) {
      const int n   = nbase + nt * 16 + col;   // output feature in [0, 3C)
      const int sec = n >> 10;                 // reference split order: k,q,v
      const int c   = n & (C_ - 1);
      const int h   = c >> 6, e = c & 63;
      const float bv = bias[n];
      const v8f av = acc[mt * 4 + nt];
#pragma unroll
      for (int v = 0; v < 8; ++v) {
        const int m  = mbase + mt * 16 + v + 8 * half;  // global token row
        const int bb = m >> 11;
        const int t  = m & (T_ - 1);
        const float val = av[v] + bv;
        const size_t bh = (size_t)bb * H_ + h;
        if (sec == 0)      Kh[(bh * T_ + t) * HD_ + e] = (bf16)val;
        else if (sec == 1) Q [(bh * T_ + t) * HD_ + e] = (bf16)val;
        else               Vt[(bh * HD_ + e) * T_ + t] = (bf16)val;
      }
    }
  }
}

// ---------------------------------------------------------------------------
// Kernel 2: flash attention. One wave per 16-query tile, streaming 32-key
// tiles (causal). V and next-K fragments issued early so the softmax VALU
// section hides their latency. grid = (B*H, T/64), block = 128.
// ---------------------------------------------------------------------------
__global__ __launch_bounds__(128) void flash_attn(
    const bf16* __restrict__ Q,   // [B*H, T, 64]
    const bf16* __restrict__ Kh,  // [B*H, T, 64]
    const bf16* __restrict__ Vt,  // [B*H, 64, T]
    bf16* __restrict__ A)         // [B*T, C]
{
  __shared__ bf16 smem[4 * 16 * 32];   // per-wave 16x32 P tile (D->A relayout)
  const int lane = threadIdx.x & 31;
  const int wave = threadIdx.x >> 5;
  bf16* P = smem + wave * (16 * 32);

  const int bh = blockIdx.x;
  const int b  = bh >> 4, h = bh & 15;
  const int q0 = (blockIdx.y * 4 + wave) * 16;
  const int half = lane >> 4, col = lane & 15;

  const bf16* Qb = Q + ((size_t)bh * T_ + q0) * HD_;
  const v16bf aq0 = load_a_frag(Qb,      HD_, lane);   // hd 0..31
  const v16bf aq1 = load_a_frag(Qb + 32, HD_, lane);   // hd 32..63

  v8f o[4] = {};
  float mrow[8], lrow[8];
#pragma unroll
  for (int v = 0; v < 8; ++v) { mrow[v] = -1e30f; lrow[v] = 0.f; }

  const bf16* Kbase = Kh + (size_t)bh * T_ * HD_;
  const bf16* Vbase = Vt + (size_t)bh * HD_ * T_;

  // preload key tile 0 fragments
  v16bf kb0 = load_b_frag(Kbase,            HD_, lane);  // keys 0..15,  hd 0..31
  v16bf kb1 = load_b_frag(Kbase + 32,       HD_, lane);  // keys 0..15,  hd 32..63
  v16bf kb2 = load_b_frag(Kbase + 16 * HD_,      HD_, lane);
  v16bf kb3 = load_b_frag(Kbase + 16 * HD_ + 32, HD_, lane);

  const int nkt = ((q0 + 15) >> 5) + 1;   // key tiles covering keys 0..q0+15
  for (int kt = 0; kt < nkt; ++kt) {
    const int k0 = kt * 32;

    // S = Q K^T  (16 x 32), 4 WMMAs
    v8f s0 = {}, s1 = {};
    s0 = wmma_bf16(aq0, kb0, s0);
    s0 = wmma_bf16(aq1, kb1, s0);
    s1 = wmma_bf16(aq0, kb2, s1);
    s1 = wmma_bf16(aq1, kb3, s1);

    // early-issue: V fragments for this tile (consumed after softmax) ...
    const bf16* Vb = Vbase + k0;
    v16bf vb0 = load_b_frag(Vb,                     T_, lane);
    v16bf vb1 = load_b_frag(Vb + (size_t)16 * T_,   T_, lane);
    v16bf vb2 = load_b_frag(Vb + (size_t)32 * T_,   T_, lane);
    v16bf vb3 = load_b_frag(Vb + (size_t)48 * T_,   T_, lane);
    // ... and next tile's K fragments
    v16bf nkb0 = kb0, nkb1 = kb1, nkb2 = kb2, nkb3 = kb3;
    if (kt + 1 < nkt) {
      const bf16* Kn = Kbase + (size_t)(k0 + 32) * HD_;
      nkb0 = load_b_frag(Kn,            HD_, lane);
      nkb1 = load_b_frag(Kn + 32,       HD_, lane);
      nkb2 = load_b_frag(Kn + 16 * HD_,      HD_, lane);
      nkb3 = load_b_frag(Kn + 16 * HD_ + 32, HD_, lane);
    }

    const bool need_mask = (k0 + 31) > q0;
#pragma unroll
    for (int v = 0; v < 8; ++v) {
      const int row = q0 + v + 8 * half;       // global query index
      float x0 = s0[v] * 0.125f;               // 1/sqrt(64)
      float x1 = s1[v] * 0.125f;
      if (need_mask) {
        if (k0 + col > row)      x0 = -1e30f;
        if (k0 + 16 + col > row) x1 = -1e30f;
      }
      float tmax = fmaxf(x0, x1);
#pragma unroll
      for (int off = 8; off; off >>= 1)
        tmax = fmaxf(tmax, __shfl_xor(tmax, off, 32));
      const float mnew  = fmaxf(mrow[v], tmax);
      const float alpha = __expf(mrow[v] - mnew);
      const float p0 = __expf(x0 - mnew);
      const float p1 = __expf(x1 - mnew);
      float ps = p0 + p1;
#pragma unroll
      for (int off = 8; off; off >>= 1) ps += __shfl_xor(ps, off, 32);
      lrow[v] = lrow[v] * alpha + ps;
      mrow[v] = mnew;
#pragma unroll
      for (int j = 0; j < 4; ++j) o[j][v] = o[j][v] * alpha;
      const int r = v + 8 * half;
      P[r * 32 + col]      = (bf16)p0;
      P[r * 32 + col + 16] = (bf16)p1;
    }
    // wave-local LDS store->load ordering
    asm volatile("s_wait_dscnt 0" ::: "memory");

    // O += P V  (16 x 64), 4 WMMAs, A-fragment (P) reused 4x
    const v16bf ap = load_a_frag(P, 32, lane);
    o[0] = wmma_bf16(ap, vb0, o[0]);
    o[1] = wmma_bf16(ap, vb1, o[1]);
    o[2] = wmma_bf16(ap, vb2, o[2]);
    o[3] = wmma_bf16(ap, vb3, o[3]);

    kb0 = nkb0; kb1 = nkb1; kb2 = nkb2; kb3 = nkb3;
  }

  // normalize and store attn output in [B*T, C] (bf16 for final GEMM)
#pragma unroll
  for (int v = 0; v < 8; ++v) {
    const float inv = 1.0f / lrow[v];
    const int t = q0 + v + 8 * half;
    bf16* dst = A + ((size_t)b * T_ + t) * C_ + h * HD_;
#pragma unroll
    for (int j = 0; j < 4; ++j) dst[j * 16 + col] = (bf16)(o[j][v] * inv);
  }
}

// ---------------------------------------------------------------------------
// Kernel 3: out = Attn @ Wo^T + b_o (fp32 output). Same 32x64 wave tiling.
//   grid = (M/256, C/64), block = 256.
// ---------------------------------------------------------------------------
__global__ __launch_bounds__(256) void out_gemm(
    const bf16* __restrict__ A,     // [B*T, C]
    const bf16* __restrict__ W,     // [C, C]
    const float* __restrict__ bias, // [C]
    float* __restrict__ out)        // [B*T, C]
{
  const int lane  = threadIdx.x & 31;
  const int wave  = threadIdx.x >> 5;
  const int mbase = blockIdx.x * 256 + wave * 32;
  const int nbase = blockIdx.y * 64;

  const bf16* Ap = A + (size_t)mbase * C_;
  const bf16* Bp = W + (size_t)nbase * C_;

  v16bf a0 = load_a_frag(Ap, C_, lane);
  v16bf a1 = load_a_frag(Ap + (size_t)16 * C_, C_, lane);
  v16bf b0 = load_b_frag(Bp, C_, lane);
  v16bf b1 = load_b_frag(Bp + (size_t)16 * C_, C_, lane);
  v16bf b2 = load_b_frag(Bp + (size_t)32 * C_, C_, lane);
  v16bf b3 = load_b_frag(Bp + (size_t)48 * C_, C_, lane);

  v8f acc[8] = {};
  for (int kk = 32; kk < C_; kk += 32) {
    v16bf na0 = load_a_frag(Ap + kk, C_, lane);
    v16bf na1 = load_a_frag(Ap + (size_t)16 * C_ + kk, C_, lane);
    v16bf nb0 = load_b_frag(Bp + kk, C_, lane);
    v16bf nb1 = load_b_frag(Bp + (size_t)16 * C_ + kk, C_, lane);
    v16bf nb2 = load_b_frag(Bp + (size_t)32 * C_ + kk, C_, lane);
    v16bf nb3 = load_b_frag(Bp + (size_t)48 * C_ + kk, C_, lane);
    if (kk + 64 < C_) {
      __builtin_prefetch((const void*)(Ap + kk + 64), 0, 0);
      __builtin_prefetch((const void*)(Bp + kk + 64), 0, 0);
    }
    acc[0] = wmma_bf16(a0, b0, acc[0]);
    acc[1] = wmma_bf16(a0, b1, acc[1]);
    acc[2] = wmma_bf16(a0, b2, acc[2]);
    acc[3] = wmma_bf16(a0, b3, acc[3]);
    acc[4] = wmma_bf16(a1, b0, acc[4]);
    acc[5] = wmma_bf16(a1, b1, acc[5]);
    acc[6] = wmma_bf16(a1, b2, acc[6]);
    acc[7] = wmma_bf16(a1, b3, acc[7]);
    a0 = na0; a1 = na1; b0 = nb0; b1 = nb1; b2 = nb2; b3 = nb3;
  }
  acc[0] = wmma_bf16(a0, b0, acc[0]);
  acc[1] = wmma_bf16(a0, b1, acc[1]);
  acc[2] = wmma_bf16(a0, b2, acc[2]);
  acc[3] = wmma_bf16(a0, b3, acc[3]);
  acc[4] = wmma_bf16(a1, b0, acc[4]);
  acc[5] = wmma_bf16(a1, b1, acc[5]);
  acc[6] = wmma_bf16(a1, b2, acc[6]);
  acc[7] = wmma_bf16(a1, b3, acc[7]);

  const int half = lane >> 4, col = lane & 15;
#pragma unroll
  for (int mt = 0; mt < 2; ++mt) {
#pragma unroll
    for (int nt = 0; nt < 4; ++nt) {
      const int n = nbase + nt * 16 + col;
      const float bv = bias[n];
      const v8f av = acc[mt * 4 + nt];
#pragma unroll
      for (int v = 0; v < 8; ++v) {
        const int m = mbase + mt * 16 + v + 8 * half;
        out[(size_t)m * C_ + n] = av[v] + bv;
      }
    }
  }
}

// ---------------------------------------------------------------------------
// Host launch
// ---------------------------------------------------------------------------
extern "C" void kernel_launch(void* const* d_in, const int* in_sizes, int n_in,
                              void* d_out, int out_size, void* d_ws, size_t ws_size,
                              hipStream_t stream) {
  const float* x      = (const float*)d_in[0];
  const float* w_attn = (const float*)d_in[1];
  const float* b_attn = (const float*)d_in[2];
  const float* w_o    = (const float*)d_in[3];
  const float* b_o    = (const float*)d_in[4];
  float* out = (float*)d_out;

  // workspace carving (~48 MB total)
  char* ws = (char*)d_ws;
  size_t off = 0;
  auto carve = [&](size_t bytes) {
    void* p = ws + off;
    off = (off + bytes + 255) & ~(size_t)255;
    return p;
  };
  bf16* Xbf  = (bf16*)carve((size_t)B_ * T_ * C_ * sizeof(bf16));
  bf16* Wbf  = (bf16*)carve((size_t)3 * C_ * C_ * sizeof(bf16));
  bf16* WObf = (bf16*)carve((size_t)C_ * C_ * sizeof(bf16));
  bf16* Q    = (bf16*)carve((size_t)B_ * H_ * T_ * HD_ * sizeof(bf16));
  bf16* Kh   = (bf16*)carve((size_t)B_ * H_ * T_ * HD_ * sizeof(bf16));
  bf16* Vt   = (bf16*)carve((size_t)B_ * H_ * T_ * HD_ * sizeof(bf16));
  bf16* Abf  = (bf16*)carve((size_t)B_ * T_ * C_ * sizeof(bf16));
  (void)ws_size; (void)in_sizes; (void)n_in; (void)out_size;

  cast_f32_bf16<<<1024, 256, 0, stream>>>(x,      Xbf,  B_ * T_ * C_ / 4);
  cast_f32_bf16<<<1024, 256, 0, stream>>>(w_attn, Wbf,  3 * C_ * C_ / 4);
  cast_f32_bf16<<<512,  256, 0, stream>>>(w_o,    WObf, C_ * C_ / 4);

  // M = B*T = 4096 -> 16 row-blocks of 256; N = 3072 -> 48 col-blocks of 64
  qkv_gemm<<<dim3(16, 48), 256, 0, stream>>>(Xbf, Wbf, b_attn, Q, Kh, Vt);

  // B*H = 32 head-slices, T/64 = 32 (4 q-tiles of 16 per block)
  flash_attn<<<dim3(32, 32), 128, 0, stream>>>(Q, Kh, Vt, Abf);

  // M = 4096 -> 16 blocks; N = 1024 -> 16 blocks
  out_gemm<<<dim3(16, 16), 256, 0, stream>>>(Abf, WObf, b_o, out);
}
